// SelfAttention_61933428408734
// MI455X (gfx1250) — matrix operand order repaired and checked by reading, not compiled, not run on hardware
//
#include <hip/hip_runtime.h>

typedef __attribute__((ext_vector_type(16))) _Float16 v16h;
typedef __attribute__((ext_vector_type(8)))  _Float16 v8h;
typedef __attribute__((ext_vector_type(8)))  float    v8f;

#define C_DIM 256
#define NH    4
#define HD    64
#define NPIX  4096
#define CPG    8   // channels per group

// ---------------------------------------------------------------------------
// WMMA helper: D = A(16x32 f16) * B(32x16 f16) + C(16x16 f32)
// ---------------------------------------------------------------------------
__device__ __forceinline__ v8f wmma16(v16h a, v16h b, v8f c) {
    return __builtin_amdgcn_wmma_f32_16x16x32_f16(
        false, a, false, b, (short)0, c, false, false);
}

// A-fragment (16x32 f16), row-major source, leading dim lda, k-offset kk.
// lane<16: row=lane, K in {kk..kk+7, kk+16..kk+23}; lane>=16: K +8.
__device__ __forceinline__ v16h load_a16(const _Float16* p, int lda, int kk) {
    int lane = threadIdx.x & 31;
    int row  = lane & 15;
    int kb   = kk + ((lane >> 4) << 3);
    v8h lo = *(const v8h*)(p + (size_t)row * lda + kb);
    v8h hi = *(const v8h*)(p + (size_t)row * lda + kb + 16);
    v16h a;
#pragma unroll
    for (int i = 0; i < 8; ++i) { a[i] = lo[i]; a[i + 8] = hi[i]; }
    return a;
}

// B-fragment (32x16 f16): element B[k][n] = p[n*ldb + k].  Lane = column n,
// lane<16: K in {kk..kk+7, kk+16..kk+23}; lane>=16: K +8.
__device__ __forceinline__ v16h load_b16(const _Float16* p, int ldb, int kk) {
    int lane = threadIdx.x & 31;
    int col  = lane & 15;
    int kb   = kk + ((lane >> 4) << 3);
    v8h lo = *(const v8h*)(p + (size_t)col * ldb + kb);
    v8h hi = *(const v8h*)(p + (size_t)col * ldb + kb + 16);
    v16h b;
#pragma unroll
    for (int i = 0; i < 8; ++i) { b[i] = lo[i]; b[i + 8] = hi[i]; }
    return b;
}

// Async global->LDS copy of 16 bytes (per lane).  Tracked by ASYNCcnt.
__device__ __forceinline__ void async_b128(_Float16* l, const _Float16* g) {
    unsigned loff = (unsigned)(unsigned long long)l;   // LDS byte address
    asm volatile("global_load_async_to_lds_b128 %0, %1, off"
                 :: "v"(loff), "v"(g) : "memory");
}

// ---------------------------------------------------------------------------
// K0: convert fp32 weights -> f16
// ---------------------------------------------------------------------------
__global__ __launch_bounds__(256) void convw_kernel(
    const float* __restrict__ qw, const float* __restrict__ ow,
    _Float16* __restrict__ qh, _Float16* __restrict__ oh) {
    int i = blockIdx.x * 256 + threadIdx.x;
    if (i < 768 * 256) qh[i] = (_Float16)qw[i];
    int j = i - 768 * 256;
    if (j >= 0 && j < 256 * 256) oh[j] = (_Float16)ow[j];
}

// ---------------------------------------------------------------------------
// K1: GroupNorm -> h as [B, N, C] f16 (B-fragment friendly layout)
// ---------------------------------------------------------------------------
__global__ __launch_bounds__(256) void gn_kernel(
    const float* __restrict__ x, const float* __restrict__ gw,
    const float* __restrict__ gb, _Float16* __restrict__ h) {
    int bid = blockIdx.x;
    int b = bid >> 5;
    int g = bid & 31;
    int t = threadIdx.x;
    const float* xb = x + ((size_t)b * C_DIM + g * CPG) * NPIX;

    float s = 0.f, ss = 0.f;
    for (int e = t; e < CPG * NPIX; e += 256) {
        float v = xb[e];
        s += v; ss += v * v;
    }
    __shared__ float r0[256], r1[256];
    r0[t] = s; r1[t] = ss;
    __syncthreads();
    for (int off = 128; off > 0; off >>= 1) {
        if (t < off) { r0[t] += r0[t + off]; r1[t] += r1[t + off]; }
        __syncthreads();
    }
    float mean = r0[0] * (1.0f / 32768.0f);
    float var  = r1[0] * (1.0f / 32768.0f) - mean * mean;
    float rstd = rsqrtf(var + 1e-5f);

    for (int e = t; e < CPG * NPIX; e += 256) {
        int c = e >> 12;
        int n = e & 4095;
        int cg = g * CPG + c;
        float v = (xb[e] - mean) * rstd * gw[cg] + gb[cg];
        h[((size_t)b * NPIX + n) * C_DIM + cg] = (_Float16)v;
    }
}

// ---------------------------------------------------------------------------
// K2: QKV GEMM.  Q,K -> [B,H,N,D]; V -> [B,H,D,N] (AV B-fragment layout)
// ---------------------------------------------------------------------------
__global__ __launch_bounds__(256) void qkv_kernel(
    const _Float16* __restrict__ wq, const _Float16* __restrict__ h,
    const float* __restrict__ bias,
    _Float16* __restrict__ q, _Float16* __restrict__ k,
    _Float16* __restrict__ v) {
    int bid = blockIdx.x;
    int mb = bid % 6;
    int nb = (bid / 6) % 64;
    int b  = bid / 384;
    int w = threadIdx.x >> 5, lane = threadIdx.x & 31;
    int m0 = mb * 128 + w * 16;
    int n0 = nb * 64;
    const _Float16* hb = h + (size_t)b * NPIX * C_DIM;

    v8f acc[4] = {v8f{}, v8f{}, v8f{}, v8f{}};
#pragma unroll 2
    for (int kk = 0; kk < C_DIM; kk += 32) {
        v16h a = load_a16(wq + (size_t)m0 * C_DIM, C_DIM, kk);
#pragma unroll
        for (int t = 0; t < 4; ++t) {
            v16h bf = load_b16(hb + (size_t)(n0 + t * 16) * C_DIM, C_DIM, kk);
            acc[t] = wmma16(a, bf, acc[t]);
        }
    }

    int col = lane & 15;
    int rb  = (lane >> 4) * 8;
#pragma unroll
    for (int t = 0; t < 4; ++t) {
#pragma unroll
        for (int i = 0; i < 8; ++i) {
            int row = m0 + rb + i;
            int n   = n0 + t * 16 + col;
            _Float16 hv = (_Float16)(acc[t][i] + bias[row]);
            if (row < 256) {
                int hd = row >> 6, d = row & 63;
                q[(((size_t)b * NH + hd) * NPIX + n) * HD + d] = hv;
            } else if (row < 512) {
                int r = row - 256, hd = r >> 6, d = r & 63;
                k[(((size_t)b * NH + hd) * NPIX + n) * HD + d] = hv;
            } else {
                int r = row - 512, hd = r >> 6, d = r & 63;
                v[(((size_t)b * NH + hd) * HD + d) * NPIX + n] = hv;
            }
        }
    }
}

// ---------------------------------------------------------------------------
// K3: flash attention with async global->LDS K/V staging (double buffered).
// Block = 8 waves = 128 query rows; 64-key chunks; per-wave 16 WMMA/chunk.
// ---------------------------------------------------------------------------
__global__ __launch_bounds__(256) void attn_kernel(
    const _Float16* __restrict__ q, const _Float16* __restrict__ k,
    const _Float16* __restrict__ v, _Float16* __restrict__ ao) {
    __shared__ __align__(128) _Float16 kst[2][64 * 64];  // [local j][d]
    __shared__ __align__(128) _Float16 vst[2][64 * 64];  // [d][local j]
    __shared__ __align__(128) _Float16 pst[8][16 * 64];  // per-wave P

    int bid = blockIdx.x;
    int it = bid & 31;
    int hh = (bid >> 5) & 3;
    int b  = bid >> 7;
    int tid = threadIdx.x;
    int w = tid >> 5, lane = tid & 31;
    int i0 = it * 128 + w * 16;

    const _Float16* qb = q + ((size_t)b * NH + hh) * NPIX * HD;
    const _Float16* kb = k + ((size_t)b * NH + hh) * NPIX * HD;
    const _Float16* vb = v + ((size_t)b * NH + hh) * HD * NPIX;

    // stage one 64-key K tile (64x64) and V tile (64x64) into LDS buffer
    int srow = tid >> 2;        // 0..63
    int sseg = tid & 3;         // 4 x 32B segments per 128B row
    auto stage = [&](int buf, int j) {
        const _Float16* gk = kb + (size_t)(j + srow) * HD + sseg * 16;
        _Float16* lk = &kst[buf][srow * 64 + sseg * 16];
        async_b128(lk, gk);
        async_b128(lk + 8, gk + 8);
        const _Float16* gv = vb + (size_t)srow * NPIX + j + sseg * 16;
        _Float16* lv = &vst[buf][srow * 64 + sseg * 16];
        async_b128(lv, gv);
        async_b128(lv + 8, gv + 8);
    };

    v16h aq0 = load_a16(qb + (size_t)i0 * HD, HD, 0);
    v16h aq1 = load_a16(qb + (size_t)i0 * HD, HD, 32);

    v8f o0 = {}, o1 = {}, o2 = {}, o3 = {};
    float m[8], l[8];
#pragma unroll
    for (int i = 0; i < 8; ++i) { m[i] = -3.0e38f; l[i] = 0.f; }

    const float scale = 0.125f;
    int col = lane & 15;
    int rb  = (lane >> 4) * 8;
    int kb2 = (lane >> 4) << 3;

    stage(0, 0);
    for (int jc = 0; jc < 64; ++jc) {
        if (jc + 1 < 64) {
            stage((jc + 1) & 1, (jc + 1) * 64);
            asm volatile("s_wait_asynccnt 0x4" ::: "memory");
        } else {
            asm volatile("s_wait_asynccnt 0x0" ::: "memory");
        }
        __syncthreads();
        const _Float16* kbuf = &kst[jc & 1][0];
        const _Float16* vbuf = &vst[jc & 1][0];

        // scores: 16 rows x 64 keys = 4 C-tiles, K over d = 64
        v8f s0 = {}, s1 = {}, s2 = {}, s3 = {};
        s0 = wmma16(aq0, load_b16(kbuf +  0 * 64, 64, 0),  s0);
        s0 = wmma16(aq1, load_b16(kbuf +  0 * 64, 64, 32), s0);
        s1 = wmma16(aq0, load_b16(kbuf + 16 * 64, 64, 0),  s1);
        s1 = wmma16(aq1, load_b16(kbuf + 16 * 64, 64, 32), s1);
        s2 = wmma16(aq0, load_b16(kbuf + 32 * 64, 64, 0),  s2);
        s2 = wmma16(aq1, load_b16(kbuf + 32 * 64, 64, 32), s2);
        s3 = wmma16(aq0, load_b16(kbuf + 48 * 64, 64, 0),  s3);
        s3 = wmma16(aq1, load_b16(kbuf + 48 * 64, 64, 32), s3);

#pragma unroll
        for (int i = 0; i < 8; ++i) {
            float a0 = s0[i] * scale, a1 = s1[i] * scale;
            float a2 = s2[i] * scale, a3 = s3[i] * scale;
            float mx = fmaxf(fmaxf(a0, a1), fmaxf(a2, a3));
            mx = fmaxf(mx, __shfl_xor(mx, 1, 32));
            mx = fmaxf(mx, __shfl_xor(mx, 2, 32));
            mx = fmaxf(mx, __shfl_xor(mx, 4, 32));
            mx = fmaxf(mx, __shfl_xor(mx, 8, 32));
            float mn = fmaxf(m[i], mx);
            float al = __expf(m[i] - mn);
            float p0 = __expf(a0 - mn), p1 = __expf(a1 - mn);
            float p2 = __expf(a2 - mn), p3 = __expf(a3 - mn);
            float rs = (p0 + p1) + (p2 + p3);
            rs += __shfl_xor(rs, 1, 32);
            rs += __shfl_xor(rs, 2, 32);
            rs += __shfl_xor(rs, 4, 32);
            rs += __shfl_xor(rs, 8, 32);
            l[i] = l[i] * al + rs;
            m[i] = mn;
            o0[i] *= al; o1[i] *= al; o2[i] *= al; o3[i] *= al;
            int row = i + rb;
            pst[w][row * 64 + col]      = (_Float16)p0;
            pst[w][row * 64 + col + 16] = (_Float16)p1;
            pst[w][row * 64 + col + 32] = (_Float16)p2;
            pst[w][row * 64 + col + 48] = (_Float16)p3;
        }
        asm volatile("s_wait_dscnt 0x0" ::: "memory");   // same-wave LDS RAW

        // P (16x64) as two A-fragments from this wave's staging buffer
        int rowA = lane & 15;
        v8h lo0 = *(const v8h*)&pst[w][rowA * 64 + kb2];
        v8h hi0 = *(const v8h*)&pst[w][rowA * 64 + kb2 + 16];
        v8h lo1 = *(const v8h*)&pst[w][rowA * 64 + 32 + kb2];
        v8h hi1 = *(const v8h*)&pst[w][rowA * 64 + 32 + kb2 + 16];
        v16h ap0, ap1;
#pragma unroll
        for (int i = 0; i < 8; ++i) {
            ap0[i] = lo0[i]; ap0[i + 8] = hi0[i];
            ap1[i] = lo1[i]; ap1[i + 8] = hi1[i];
        }

        // O += P @ V^T   (V^T fragments: lane = d column, K = local j)
        o0 = wmma16(ap0, load_b16(vbuf +  0 * 64, 64, 0),  o0);
        o0 = wmma16(ap1, load_b16(vbuf +  0 * 64, 64, 32), o0);
        o1 = wmma16(ap0, load_b16(vbuf + 16 * 64, 64, 0),  o1);
        o1 = wmma16(ap1, load_b16(vbuf + 16 * 64, 64, 32), o1);
        o2 = wmma16(ap0, load_b16(vbuf + 32 * 64, 64, 0),  o2);
        o2 = wmma16(ap1, load_b16(vbuf + 32 * 64, 64, 32), o2);
        o3 = wmma16(ap0, load_b16(vbuf + 48 * 64, 64, 0),  o3);
        o3 = wmma16(ap1, load_b16(vbuf + 48 * 64, 64, 32), o3);

        __syncthreads();   // all waves done reading before buffer reuse
    }

    // normalize, store O^T as [B, N, C] f16 (C = head*64 + d)
    _Float16* aob = ao + (size_t)b * NPIX * C_DIM + hh * HD;
#pragma unroll
    for (int i = 0; i < 8; ++i) {
        float inv = 1.0f / l[i];
        int n = i0 + rb + i;
        _Float16* r = aob + (size_t)n * C_DIM;
        r[ 0 + col] = (_Float16)(o0[i] * inv);
        r[16 + col] = (_Float16)(o1[i] * inv);
        r[32 + col] = (_Float16)(o2[i] * inv);
        r[48 + col] = (_Float16)(o3[i] * inv);
    }
}

// ---------------------------------------------------------------------------
// K4: output projection + bias + residual (fp32 out)
// ---------------------------------------------------------------------------
__global__ __launch_bounds__(256) void proj_kernel(
    const _Float16* __restrict__ wo, const _Float16* __restrict__ ao,
    const float* __restrict__ ob, const float* __restrict__ x,
    float* __restrict__ y) {
    int bid = blockIdx.x;
    int mb = bid & 1;
    int nb = (bid >> 1) & 63;
    int b  = bid >> 7;
    int w = threadIdx.x >> 5, lane = threadIdx.x & 31;
    int m0 = mb * 128 + w * 16;
    int n0 = nb * 64;
    const _Float16* aob = ao + (size_t)b * NPIX * C_DIM;

    v8f acc[4] = {v8f{}, v8f{}, v8f{}, v8f{}};
#pragma unroll 2
    for (int kk = 0; kk < C_DIM; kk += 32) {
        v16h a = load_a16(wo + (size_t)m0 * C_DIM, C_DIM, kk);
#pragma unroll
        for (int t = 0; t < 4; ++t) {
            v16h bf = load_b16(aob + (size_t)(n0 + t * 16) * C_DIM, C_DIM, kk);
            acc[t] = wmma16(a, bf, acc[t]);
        }
    }

    int col = lane & 15;
    int rb  = (lane >> 4) * 8;
#pragma unroll
    for (int t = 0; t < 4; ++t) {
#pragma unroll
        for (int i = 0; i < 8; ++i) {
            int row = m0 + rb + i;
            int n   = n0 + t * 16 + col;
            size_t idx = ((size_t)b * C_DIM + row) * NPIX + n;
            y[idx] = acc[t][i] + ob[row] + x[idx];
        }
    }
}

// ---------------------------------------------------------------------------
// Launch
// ---------------------------------------------------------------------------
extern "C" void kernel_launch(void* const* d_in, const int* in_sizes, int n_in,
                              void* d_out, int out_size, void* d_ws, size_t ws_size,
                              hipStream_t stream) {
    const float* x     = (const float*)d_in[0];
    const float* gn_w  = (const float*)d_in[1];
    const float* gn_b  = (const float*)d_in[2];
    const float* qkv_w = (const float*)d_in[3];
    const float* qkv_b = (const float*)d_in[4];
    const float* out_w = (const float*)d_in[5];
    const float* out_b = (const float*)d_in[6];
    float* y = (float*)d_out;

    char* ws = (char*)d_ws;
    const size_t MB8 = (size_t)8 << 20;
    _Float16* h   = (_Float16*)(ws);
    _Float16* q   = (_Float16*)(ws + 1 * MB8);
    _Float16* k   = (_Float16*)(ws + 2 * MB8);
    _Float16* v   = (_Float16*)(ws + 3 * MB8);
    _Float16* ao  = (_Float16*)(ws + 4 * MB8);
    _Float16* wqh = (_Float16*)(ws + 5 * MB8);
    _Float16* woh = (_Float16*)(ws + 5 * MB8 + (size_t)768 * 256 * 2);

    convw_kernel<<<1024, 256, 0, stream>>>(qkv_w, out_w, wqh, woh);
    gn_kernel<<<128, 256, 0, stream>>>(x, gn_w, gn_b, h);
    qkv_kernel<<<1536, 256, 0, stream>>>(wqh, h, qkv_b, q, k, v);
    attn_kernel<<<512, 256, 0, stream>>>(q, k, v, ao);
    proj_kernel<<<512, 256, 0, stream>>>(woh, ao, out_b, x, y);
}
